// PCT_63170378990045
// MI455X (gfx1250) — matrix operand order repaired and checked by reading, not compiled, not run on hardware
//
#include <hip/hip_runtime.h>
#include <hip/hip_bf16.h>
#include <math.h>

// ---------------------------------------------------------------------------
// Problem constants (match reference)
// ---------------------------------------------------------------------------
#define BB 2
#define NN 8192
#define CI 64
#define CP 64
#define HH 128
#define WW 128
#define DD 32            // qk channels
#define BN_EPS 1e-5f

typedef _Float16 half_t;
typedef __attribute__((ext_vector_type(16))) _Float16 v16h;
typedef __attribute__((ext_vector_type(8)))  _Float16 v8h;
typedef __attribute__((ext_vector_type(8)))  float    v8f;

// 16-lane-group reductions (wave32; xor masks <=8 stay inside each half)
__device__ __forceinline__ float grp_max(float v) {
    v = fmaxf(v, __shfl_xor(v, 1, 32));
    v = fmaxf(v, __shfl_xor(v, 2, 32));
    v = fmaxf(v, __shfl_xor(v, 4, 32));
    v = fmaxf(v, __shfl_xor(v, 8, 32));
    return v;
}
__device__ __forceinline__ float grp_sum(float v) {
    v += __shfl_xor(v, 1, 32);
    v += __shfl_xor(v, 2, 32);
    v += __shfl_xor(v, 4, 32);
    v += __shfl_xor(v, 8, 32);
    return v;
}

// load a 16-bit B fragment: lane holds column (base+hl), 16 contiguous K
__device__ __forceinline__ v16h load_bfrag(const half_t* __restrict__ rows,
                                           int stride, int col, int kbase) {
    const half_t* rp = rows + (size_t)col * stride + kbase;
    v8h lo = *(const v8h*)rp;
    v8h hp = *(const v8h*)(rp + 8);
    v16h b;
#pragma unroll
    for (int i = 0; i < 8; ++i) { b[i] = lo[i]; b[i + 8] = hp[i]; }
    return b;
}

// ---------------------------------------------------------------------------
// Kernel 0: zero BN accumulators (128 floats)
// ---------------------------------------------------------------------------
__global__ void k_zero(float* __restrict__ bnacc) {
    if (threadIdx.x < 128) bnacc[threadIdx.x] = 0.0f;
}

// ---------------------------------------------------------------------------
// Kernel 1: grid-sample + fuse + QK (f16, [B,N,32]) + V (f16, [B,64,N])
// one wave per point; 8 points per 256-thread block
// ---------------------------------------------------------------------------
__global__ void k_fuse(const float* __restrict__ pts_img,
                       const float* __restrict__ img_feats,
                       const float* __restrict__ pts_feats,
                       const float* __restrict__ qk_w,
                       const float* __restrict__ v_w,
                       const float* __restrict__ v_b,
                       half_t* __restrict__ qk_out,
                       half_t* __restrict__ v_out) {
    __shared__ float fus[8][130];          // 128 fused channels, padded
    const int lane = threadIdx.x & 31;
    const int wv   = threadIdx.x >> 5;
    const int p    = blockIdx.x * 8 + wv;  // 0 .. B*N-1
    const int b    = p >> 13;
    const int n    = p & (NN - 1);

    const float px = pts_img[(size_t)p * 2 + 0];
    const float py = pts_img[(size_t)p * 2 + 1];
    const float ix = (px + 1.0f) * 0.5f * (float)WW - 0.5f;
    const float iy = (py + 1.0f) * 0.5f * (float)HH - 0.5f;
    const float x0f = floorf(ix), y0f = floorf(iy);
    const int   x0 = (int)x0f, y0 = (int)y0f;
    const int   x1 = x0 + 1,   y1 = y0 + 1;
    const float wx1 = ix - x0f, wx0 = 1.0f - wx1;
    const float wy1 = iy - y0f, wy0 = 1.0f - wy1;
    const float vx0 = (x0 >= 0 && x0 < WW) ? 1.0f : 0.0f;
    const float vx1 = (x1 >= 0 && x1 < WW) ? 1.0f : 0.0f;
    const float vy0 = (y0 >= 0 && y0 < HH) ? 1.0f : 0.0f;
    const float vy1 = (y1 >= 0 && y1 < HH) ? 1.0f : 0.0f;
    const int xc0 = x0 < 0 ? 0 : (x0 > WW - 1 ? WW - 1 : x0);
    const int xc1 = x1 < 0 ? 0 : (x1 > WW - 1 ? WW - 1 : x1);
    const int yc0 = y0 < 0 ? 0 : (y0 > HH - 1 ? HH - 1 : y0);
    const int yc1 = y1 < 0 ? 0 : (y1 > HH - 1 ? HH - 1 : y1);
    const float w00 = wy0 * wx0 * vy0 * vx0;
    const float w01 = wy0 * wx1 * vy0 * vx1;
    const float w10 = wy1 * wx0 * vy1 * vx0;
    const float w11 = wy1 * wx1 * vy1 * vx1;

    const float* imgb = img_feats + (size_t)b * CI * HH * WW;
#pragma unroll
    for (int cc = 0; cc < 2; ++cc) {
        const int c = lane + cc * 32;
        const float* ch = imgb + (size_t)c * (HH * WW);
        float g = ch[yc0 * WW + xc0] * w00 + ch[yc0 * WW + xc1] * w01 +
                  ch[yc1 * WW + xc0] * w10 + ch[yc1 * WW + xc1] * w11;
        fus[wv][c]      = g;
        fus[wv][64 + c] = pts_feats[((size_t)b * CP + c) * NN + n];
    }
    __syncthreads();

    // QK: d = lane (32 outputs), sum over 128 fused channels
    float aq = 0.0f;
#pragma unroll 8
    for (int c = 0; c < 128; ++c) aq += qk_w[lane * 128 + c] * fus[wv][c];
    qk_out[(size_t)p * DD + lane] = (half_t)aq;

    // V: two outputs per lane, sum over 64 point channels
#pragma unroll
    for (int cc = 0; cc < 2; ++cc) {
        const int o = lane + cc * 32;
        float av = v_b[o];
#pragma unroll 8
        for (int c = 0; c < 64; ++c) av += v_w[o * 64 + c] * fus[wv][64 + c];
        v_out[((size_t)b * CP + o) * NN + n] = (half_t)av;
    }
}

// ---------------------------------------------------------------------------
// Kernel 2: flash pass 1 — per-row softmax stats mx[n], sm[n]
// one wave per 16-row tile of n.  Two WMMA sweeps:
//   sweep A: lane-local elementwise running max (no shuffles in loop)
//   sweep B: recompute E, lane-local sum of exp(E - mx)
// cross-lane reduction happens exactly once per sweep.
// ---------------------------------------------------------------------------
__global__ void k_rowstats(const half_t* __restrict__ qk,
                           float* __restrict__ mx, float* __restrict__ sm) {
    const int lane = threadIdx.x & 31;
    const int wv   = threadIdx.x >> 5;
    const int t    = blockIdx.x * 8 + wv;          // 0 .. B*N/16-1
    const int b    = t >> 9;                       // 512 tiles per batch
    const int n0   = (t & 511) << 4;
    const half_t* qkb = qk + (size_t)b * NN * DD;
    const int hl = lane & 15, hi = lane >> 4;

    // A fragment: rows n0..n0+15, this code's k order (kbase = hi*16, 16 contiguous)
    v16h a;
    {
        const half_t* rp = qkb + (size_t)(n0 + hl) * DD + hi * 16;
        v8h lo = *(const v8h*)rp;
        v8h hp = *(const v8h*)(rp + 8);
#pragma unroll
        for (int i = 0; i < 8; ++i) { a[i] = lo[i]; a[i + 8] = hp[i]; }
    }

    // ---- sweep A: row max (lane-local partials) ----
    float vmax8[8];
#pragma unroll
    for (int r = 0; r < 8; ++r) vmax8[r] = -INFINITY;

#pragma unroll 2
    for (int j = 0; j < NN / 16; ++j) {
        const int m0 = j << 4;
        __builtin_prefetch(qkb + (size_t)(m0 + 16 + hl) * DD, 0, 1);
        v16h bf = load_bfrag(qkb, DD, m0 + hl, hi * 16);
        v8f cz = {};
        v8f e = __builtin_amdgcn_wmma_f32_16x16x32_f16(
            false, a, false, bf, (short)0, cz, false, false);
#pragma unroll
        for (int r = 0; r < 8; ++r) vmax8[r] = fmaxf(vmax8[r], e[r]);
    }
    float mx8[8];
#pragma unroll
    for (int r = 0; r < 8; ++r) mx8[r] = grp_max(vmax8[r]);

    // ---- sweep B: sum of exp(E - mx) (lane-local partials) ----
    float sp8[8];
#pragma unroll
    for (int r = 0; r < 8; ++r) sp8[r] = 0.0f;

#pragma unroll 2
    for (int j = 0; j < NN / 16; ++j) {
        const int m0 = j << 4;
        v16h bf = load_bfrag(qkb, DD, m0 + hl, hi * 16);
        v8f cz = {};
        v8f e = __builtin_amdgcn_wmma_f32_16x16x32_f16(
            false, a, false, bf, (short)0, cz, false, false);
#pragma unroll
        for (int r = 0; r < 8; ++r) sp8[r] += __expf(e[r] - mx8[r]);
    }

    if (true) {
#pragma unroll
        for (int r = 0; r < 8; ++r) sp8[r] = grp_sum(sp8[r]);
    }
    if (hl == 0) {
#pragma unroll
        for (int r = 0; r < 8; ++r) {
            const int row = n0 + r + hi * 8;
            mx[b * NN + row] = mx8[r];
            sm[b * NN + row] = sp8[r];
        }
    }
}

// ---------------------------------------------------------------------------
// Kernel 3: flash pass 2 — x_r^T[m][c] = (P^T · V^T) / (eps + rowsum(P^T))
// one wave per 16-row tile of m; stream n in chunks of 32
// ---------------------------------------------------------------------------
__global__ void k_attn(const half_t* __restrict__ qk,
                       const half_t* __restrict__ vmat,
                       const float* __restrict__ mx, const float* __restrict__ sm,
                       float* __restrict__ xrt) {
    __shared__ alignas(16) half_t stg[8][16 * 32];   // per-wave P^T staging
    const int lane = threadIdx.x & 31;
    const int wv   = threadIdx.x >> 5;
    const int t    = blockIdx.x * 8 + wv;
    const int b    = t >> 9;
    const int m0   = (t & 511) << 4;
    const half_t* qkb = qk + (size_t)b * NN * DD;
    const half_t* vb  = vmat + (size_t)b * CP * NN;
    const float*  mxb = mx + b * NN;
    const float*  smb = sm + b * NN;
    const int hl = lane & 15, hi = lane >> 4;
    half_t* st = stg[wv];

    // constant A fragment: the 16 m-rows of QK
    v16h am;
    {
        const half_t* rp = qkb + (size_t)(m0 + hl) * DD + hi * 16;
        v8h lo = *(const v8h*)rp;
        v8h hp = *(const v8h*)(rp + 8);
#pragma unroll
        for (int i = 0; i < 8; ++i) { am[i] = lo[i]; am[i + 8] = hp[i]; }
    }

    v8f acc[4];
#pragma unroll
    for (int ct = 0; ct < 4; ++ct) acc[ct] = (v8f){};
    float cs[8];
#pragma unroll
    for (int r = 0; r < 8; ++r) cs[r] = 0.0f;

    for (int nch = 0; nch < NN / 32; ++nch) {
        const int n0 = nch << 5;
        __builtin_prefetch(qkb + (size_t)(n0 + 32 + hl) * DD, 0, 1);

        // build P^T block [16 m x 32 n] in LDS, f16
#pragma unroll
        for (int hh = 0; hh < 2; ++hh) {
            const int base = n0 + hh * 16;
            v16h bf = load_bfrag(qkb, DD, base + hl, hi * 16);
            v8f cz = {};
            v8f e = __builtin_amdgcn_wmma_f32_16x16x32_f16(
                false, am, false, bf, (short)0, cz, false, false);
            const int ncol = base + hl;          // this lane's column n
            const float mxv = mxb[ncol];
            const float inv = 1.0f / smb[ncol];  // sm >= 1 always
#pragma unroll
            for (int r = 0; r < 8; ++r) {
                const float pv = __expf(e[r] - mxv) * inv;
                cs[r] += pv;                                   // partial rowsum
                st[(r + hi * 8) * 32 + hh * 16 + hl] = (half_t)pv;
            }
        }
        asm volatile("s_wait_dscnt 0" ::: "memory");  // LDS RAW within wave

        // reload as A fragment [16 m x 32 n]
        v16h ap;
        {
            const half_t* rp = st + hl * 32 + hi * 16;
            v8h lo = *(const v8h*)rp;
            v8h hp = *(const v8h*)(rp + 8);
#pragma unroll
            for (int i = 0; i < 8; ++i) { ap[i] = lo[i]; ap[i + 8] = hp[i]; }
        }
        // accumulate P^T (16x32) x V^T (32 x 64) over 4 c-tiles
#pragma unroll
        for (int ct = 0; ct < 4; ++ct) {
            v16h bv = load_bfrag(vb, NN, ct * 16 + hl, n0 + hi * 16);
            acc[ct] = __builtin_amdgcn_wmma_f32_16x16x32_f16(
                false, ap, false, bv, (short)0, acc[ct], false, false);
        }
    }

    // finalize: L1 renorm by column-sum (row-sum of P^T), write x_r^T
#pragma unroll
    for (int r = 0; r < 8; ++r) cs[r] = grp_sum(cs[r]);
#pragma unroll
    for (int ct = 0; ct < 4; ++ct) {
        const int c = ct * 16 + hl;
#pragma unroll
        for (int r = 0; r < 8; ++r) {
            const int row = m0 + r + hi * 8;
            xrt[((size_t)b * NN + row) * CP + c] = acc[ct][r] / (1e-9f + cs[r]);
        }
    }
}

// ---------------------------------------------------------------------------
// Kernel 4: y = t_w (pts_feats - x_r) + t_b ; accumulate BN sum/sumsq
// one wave per point; per-block LDS reduction, one global atomic per channel
// ---------------------------------------------------------------------------
__global__ void k_trans(const float* __restrict__ pts_feats,
                        const float* __restrict__ xrt,
                        const float* __restrict__ t_w,
                        const float* __restrict__ t_b,
                        float* __restrict__ yout, float* __restrict__ bnacc) {
    __shared__ float z[8][66];
    __shared__ float bs[64], bq[64];
    const int lane = threadIdx.x & 31;
    const int wv   = threadIdx.x >> 5;
    const int tid  = threadIdx.x;
    if (tid < 64) { bs[tid] = 0.0f; bq[tid] = 0.0f; }
    __syncthreads();

    const int p = blockIdx.x * 8 + wv;
    const int b = p >> 13;
    const int m = p & (NN - 1);
#pragma unroll
    for (int cc = 0; cc < 2; ++cc) {
        const int c = lane + cc * 32;
        z[wv][c] = pts_feats[((size_t)b * CP + c) * NN + m]
                 - xrt[(size_t)p * CP + c];
    }
    __syncthreads();
#pragma unroll
    for (int cc = 0; cc < 2; ++cc) {
        const int o = lane + cc * 32;
        float a = t_b[o];
#pragma unroll 8
        for (int c = 0; c < 64; ++c) a += t_w[o * 64 + c] * z[wv][c];
        yout[((size_t)b * CP + o) * NN + m] = a;
        atomicAdd(&bs[o], a);
        atomicAdd(&bq[o], a * a);
    }
    __syncthreads();
    if (tid < 64) {
        atomicAdd(&bnacc[tid],      bs[tid]);
        atomicAdd(&bnacc[64 + tid], bq[tid]);
    }
}

// ---------------------------------------------------------------------------
// Kernel 5: BN finalize -> per-channel scale/shift
// ---------------------------------------------------------------------------
__global__ void k_bnfin(const float* __restrict__ bnacc,
                        const float* __restrict__ gamma,
                        const float* __restrict__ beta,
                        float* __restrict__ ss) {
    const int c = threadIdx.x;
    const float inv  = 1.0f / (float)(BB * NN);
    const float mean = bnacc[c] * inv;
    const float var  = bnacc[64 + c] * inv - mean * mean;
    const float sc   = gamma[c] * rsqrtf(var + BN_EPS);
    ss[c]      = sc;
    ss[64 + c] = beta[c] - mean * sc;
}

// ---------------------------------------------------------------------------
// Kernel 6: out = pts_feats + relu(y*scale + shift)
// ---------------------------------------------------------------------------
__global__ void k_out(const float* __restrict__ y, const float* __restrict__ ss,
                      const float* __restrict__ pts_feats, float* __restrict__ out) {
    const int idx = blockIdx.x * 256 + threadIdx.x;
    const int c   = (idx >> 13) & 63;
    const float t = y[idx] * ss[c] + ss[64 + c];
    out[idx] = pts_feats[idx] + fmaxf(t, 0.0f);
}

// ---------------------------------------------------------------------------
extern "C" void kernel_launch(void* const* d_in, const int* in_sizes, int n_in,
                              void* d_out, int out_size, void* d_ws, size_t ws_size,
                              hipStream_t stream) {
    const float* pts_img   = (const float*)d_in[0];
    const float* img_feats = (const float*)d_in[1];
    const float* pts_feats = (const float*)d_in[2];
    const float* qk_w      = (const float*)d_in[3];
    const float* v_w       = (const float*)d_in[4];
    const float* v_b       = (const float*)d_in[5];
    const float* t_w       = (const float*)d_in[6];
    const float* t_b       = (const float*)d_in[7];
    const float* bn_gamma  = (const float*)d_in[8];
    const float* bn_beta   = (const float*)d_in[9];
    float* out = (float*)d_out;

    char* ws = (char*)d_ws;
    half_t* qkh  = (half_t*)(ws);                                       // 1 MB
    half_t* vh   = (half_t*)(ws + (1u << 20));                          // 2 MB
    float*  mx   = (float*)(ws + (3u << 20));                           // 64 KB
    float*  sm   = (float*)(ws + (3u << 20) + (1u << 16));              // 64 KB
    float*  xrt  = (float*)(ws + (3u << 20) + (2u << 16));              // 4 MB
    float*  yb   = (float*)(ws + (3u << 20) + (2u << 16) + (4u << 20)); // 4 MB
    float*  bnac = (float*)(ws + (3u << 20) + (2u << 16) + (8u << 20)); // 128 f
    float*  ss   = bnac + 128;                                          // 128 f

    k_zero<<<1, 128, 0, stream>>>(bnac);
    k_fuse<<<(BB * NN) / 8, 256, 0, stream>>>(pts_img, img_feats, pts_feats,
                                              qk_w, v_w, v_b, qkh, vh);
    k_rowstats<<<(BB * NN / 16) / 8, 256, 0, stream>>>(qkh, mx, sm);
    k_attn<<<(BB * NN / 16) / 8, 256, 0, stream>>>(qkh, vh, mx, sm, xrt);
    k_trans<<<(BB * NN) / 8, 256, 0, stream>>>(pts_feats, xrt, t_w, t_b, yb, bnac);
    k_bnfin<<<1, 64, 0, stream>>>(bnac, bn_gamma, bn_beta, ss);
    k_out<<<(BB * CP * NN) / 256, 256, 0, stream>>>(yb, ss, pts_feats, out);
}
